// Attention_52621939311076
// MI455X (gfx1250) — matrix operand order repaired and checked by reading, not compiled, not run on hardware
//
#include <hip/hip_runtime.h>
#include <hip/hip_bf16.h>

typedef __bf16 bf16;
typedef __bf16 v4bf  __attribute__((ext_vector_type(4)));
typedef __bf16 v8bf  __attribute__((ext_vector_type(8)));
typedef __bf16 v16bf __attribute__((ext_vector_type(16)));
typedef float  v8f   __attribute__((ext_vector_type(8)));

#define S_LEN 2048
#define D_DIM 5120
#define H_Q   64
#define H_KV  8
#define HD_DIM 128
#define QD   (H_Q * HD_DIM)    // 8192
#define KVD  (H_KV * HD_DIM)   // 1024

// ---------------- fragment helpers (ISA 7.12.2 layouts) ----------------
__device__ __forceinline__ v16bf frag_cat(v8bf lo, v8bf hi) {
    return __builtin_shufflevector(lo, hi, 0,1,2,3,4,5,6,7,8,9,10,11,12,13,14,15);
}

// A-matrix 16x32 bf16: lane%16 = row M; lanes<16 hold K={0..7,16..23},
// lanes>=16 hold K={8..15,24..31}.  'base' points at (row0, k0); stride in elems.
__device__ __forceinline__ v16bf load_frag_a(const bf16* base, int stride) {
    const int lane = threadIdx.x & 31;
    const bf16* p = base + (size_t)(lane & 15) * stride + ((lane >> 4) << 3);
    v8bf lo = *(const v8bf*)p;
    v8bf hi = *(const v8bf*)(p + 16);
    return frag_cat(lo, hi);
}

// B-matrix 32x16 bf16: lane%16 = col N; lanes<16 hold K={0..7,16..23}, lanes>=16 K={8..15,24..31}.
// 'p' = this lane's first K element (k0 + 8*(lane/16)).
__device__ __forceinline__ v16bf load_frag_b16(const bf16* p) {
    v8bf lo = *(const v8bf*)p;
    v8bf hi = *(const v8bf*)(p + 16);
    return frag_cat(lo, hi);
}

// ---------------- CDNA5 async global->LDS copy (ASYNCcnt-tracked) ----------------
__device__ __forceinline__ void async_copy_b128(const bf16* g, unsigned lds_off) {
    asm volatile("global_load_async_to_lds_b128 %0, %1, off"
                 :: "v"(lds_off), "v"(g) : "memory");
}
__device__ __forceinline__ void wait_async4() {
    asm volatile("s_wait_asynccnt 0x4" ::: "memory");
}
__device__ __forceinline__ void wait_async0() {
    asm volatile("s_wait_asynccnt 0x0" ::: "memory");
}

// ---------------- fp32 -> bf16 convert (vectorized x4) ----------------
__global__ void k_cvt_bf16(const float* __restrict__ s, bf16* __restrict__ d, size_t n4) {
    size_t i = (size_t)blockIdx.x * blockDim.x + threadIdx.x;
    if (i < n4) {
        float4 f = ((const float4*)s)[i];
        v4bf o = { (bf16)f.x, (bf16)f.y, (bf16)f.z, (bf16)f.w };
        ((v4bf*)d)[i] = o;
    }
}

// ---------------- GEMM: C[M,N] = A[M,K] * W[N,K]^T (bf16 in, f32 out) ----------------
// 128x128 block, BK=32, 8 waves; wave tile 32x64 (2 A-frags x 4 B-frags = 8 WMMA/step);
// double-buffered LDS filled by async global->LDS copies, s_wait_asynccnt pipelining.
#define BMg 128
#define BNg 128
#define BKg 32
#define LDT 40                  // padded LDS row stride (80B, keeps 16B alignment)
#define ATILE (BMg * LDT)       // elems
#define BTILE (BNg * LDT)

__global__ __launch_bounds__(256) void k_gemm_bf16(const bf16* __restrict__ A,
                                                   const bf16* __restrict__ W,
                                                   float* __restrict__ C,
                                                   int M, int N, int K) {
    __shared__ bf16 smem[2 * (ATILE + BTILE)];
    const unsigned smem_base = (unsigned)(size_t)(void*)&smem[0];   // LDS offset = addr[31:0]
    const int t    = threadIdx.x;
    const int lane = t & 31;
    const int wave = t >> 5;      // 0..7
    const int wm   = wave & 3;    // 4 m-groups of 32 rows
    const int wn   = wave >> 2;   // 2 n-groups of 64 cols
    const int m0   = blockIdx.y * BMg;
    const int n0   = blockIdx.x * BNg;

    // each thread stages 2 b128 chunks of A and 2 of B per K-step
    const int id0 = t * 2, id1 = t * 2 + 1;
    const int ar0 = id0 >> 2, ac0 = (id0 & 3) << 3;
    const int ar1 = id1 >> 2, ac1 = (id1 & 3) << 3;
    const bf16* gA0 = A + (size_t)(m0 + ar0) * K + ac0;
    const bf16* gA1 = A + (size_t)(m0 + ar1) * K + ac1;
    const bf16* gB0 = W + (size_t)(n0 + ar0) * K + ac0;
    const bf16* gB1 = W + (size_t)(n0 + ar1) * K + ac1;
    const unsigned lA0 = smem_base + (unsigned)(ar0 * LDT + ac0) * 2;
    const unsigned lA1 = smem_base + (unsigned)(ar1 * LDT + ac1) * 2;
    const unsigned lB0 = smem_base + (unsigned)(ATILE + ar0 * LDT + ac0) * 2;
    const unsigned lB1 = smem_base + (unsigned)(ATILE + ar1 * LDT + ac1) * 2;
    const unsigned bufstep = (unsigned)(ATILE + BTILE) * 2;

    v8f acc[2][4];
#pragma unroll
    for (int mi = 0; mi < 2; ++mi)
#pragma unroll
        for (int nt = 0; nt < 4; ++nt) { v8f z = {}; acc[mi][nt] = z; }

    const int nk = K / BKg;
    // preload buffer 0
    async_copy_b128(gA0, lA0); async_copy_b128(gA1, lA1);
    async_copy_b128(gB0, lB0); async_copy_b128(gB1, lB1);

    for (int i = 0; i < nk; ++i) {
        if (i + 1 < nk) {       // stage next tile into the other buffer, then drain current
            const int ke = (i + 1) * BKg;
            const unsigned bo = ((i + 1) & 1) ? bufstep : 0u;
            async_copy_b128(gA0 + ke, lA0 + bo); async_copy_b128(gA1 + ke, lA1 + bo);
            async_copy_b128(gB0 + ke, lB0 + bo); async_copy_b128(gB1 + ke, lB1 + bo);
            wait_async4();      // loads complete in order: <=4 left => current buf done
        } else {
            wait_async0();
        }
        __syncthreads();        // all waves' current buffer visible

        const bf16* sA = smem + (size_t)((i & 1) ? (ATILE + BTILE) : 0);
        const bf16* sB = sA + ATILE;
        v16bf af0 = load_frag_a(sA + (size_t)(wm * 32) * LDT, LDT);
        v16bf af1 = load_frag_a(sA + (size_t)(wm * 32 + 16) * LDT, LDT);
        v16bf bq[4];
#pragma unroll
        for (int nt = 0; nt < 4; ++nt)
            bq[nt] = load_frag_b16(sB + (size_t)(wn * 64 + nt * 16 + (lane & 15)) * LDT
                                      + ((lane >> 4) << 3));
#pragma unroll
        for (int nt = 0; nt < 4; ++nt) {
            acc[0][nt] = __builtin_amdgcn_wmma_f32_16x16x32_bf16(false, af0, false, bq[nt],
                                                                 (short)0, acc[0][nt], false, false);
            acc[1][nt] = __builtin_amdgcn_wmma_f32_16x16x32_bf16(false, af1, false, bq[nt],
                                                                 (short)0, acc[1][nt], false, false);
        }
        __syncthreads();        // WAR: next iter's async writes may now land
    }

    const int row0 = m0 + wm * 32 + ((lane >> 4) << 3);
    const int col0 = n0 + wn * 64 + (lane & 15);
#pragma unroll
    for (int mi = 0; mi < 2; ++mi)
#pragma unroll
        for (int nt = 0; nt < 4; ++nt)
#pragma unroll
            for (int v = 0; v < 8; ++v)
                C[(size_t)(row0 + mi * 16 + v) * N + col0 + nt * 16] = acc[mi][nt][v];
}

// ---------------- fused RMSNorm + RoPE, one wave per (s,head) row of 128 ----------------
__global__ __launch_bounds__(256) void k_norm_rope(const float* __restrict__ X,
                                                   const float* __restrict__ w,
                                                   const float* __restrict__ fc,
                                                   const float* __restrict__ fs,
                                                   bf16* __restrict__ Y, int NH) {
    const int lane = threadIdx.x & 31;
    const int row  = blockIdx.x * 8 + (threadIdx.x >> 5);  // s*NH + h
    const int s    = row / NH;
    const float4 xv = *(const float4*)(X + (size_t)row * HD_DIM + lane * 4);
    float ss = xv.x * xv.x + xv.y * xv.y + xv.z * xv.z + xv.w * xv.w;
#pragma unroll
    for (int o = 16; o >= 1; o >>= 1) ss += __shfl_xor(ss, o, 32);
    const float rn = rsqrtf(ss * (1.0f / HD_DIM) + 1e-6f);
    const float* wp = w + lane * 4;
    const int p0 = lane * 2, p1 = lane * 2 + 1;
    const float c0 = fc[s * 64 + p0], sn0 = fs[s * 64 + p0];
    const float c1 = fc[s * 64 + p1], sn1 = fs[s * 64 + p1];
    bf16* yp = Y + (size_t)row * HD_DIM + lane * 4;
    float ve = xv.x * rn * wp[0], vo = xv.y * rn * wp[1];
    yp[0] = (bf16)(ve * c0 - vo * sn0);
    yp[1] = (bf16)(ve * sn0 + vo * c0);
    ve = xv.z * rn * wp[2]; vo = xv.w * rn * wp[3];
    yp[2] = (bf16)(ve * c1 - vo * sn1);
    yp[3] = (bf16)(ve * sn1 + vo * c1);
}

// ---------------- V: (S, KVH, HD) f32 -> (KVH*HD, S) bf16 transpose ----------------
__global__ void k_v_transpose(const float* __restrict__ V, bf16* __restrict__ Vt) {
    size_t i = (size_t)blockIdx.x * blockDim.x + threadIdx.x;
    const size_t n = (size_t)S_LEN * KVD;
    if (i >= n) return;
    const int s = (int)(i / KVD);
    const int r = (int)(i % KVD);                 // kvh*HD + hd
    Vt[(size_t)r * S_LEN + s] = (bf16)V[i];
}

// ---------------- flash attention: grid (S/64, H), 4 waves, 16 q-rows each ----------------
__global__ __launch_bounds__(128) void k_attn(const bf16* __restrict__ Q,
                                              const bf16* __restrict__ Kk,
                                              const bf16* __restrict__ Vt,
                                              bf16* __restrict__ O) {
    __shared__ bf16 P[4][16 * 32];
    const int lane = threadIdx.x & 31;
    const int wave = threadIdx.x >> 5;
    const int h    = blockIdx.y;
    const int kvh  = h >> 3;                  // h / NREP
    const int qblk = blockIdx.x * 64;
    const int qbase = qblk + wave * 16;
    const float scale = 0.08838834764831845f; // 1/sqrt(128)

    v16bf qf[4];
    const bf16* qptr = Q + ((size_t)(qbase + (lane & 15)) * H_Q + h) * HD_DIM + ((lane >> 4) << 3);
#pragma unroll
    for (int c = 0; c < 4; ++c) {
        v8bf lo = *(const v8bf*)(qptr + c * 32);
        v8bf hi = *(const v8bf*)(qptr + c * 32 + 16);
        qf[c] = frag_cat(lo, hi);
    }

    v8f o[8];
    float mrun[8], lrun[8];
#pragma unroll
    for (int d = 0; d < 8; ++d) { v8f z = {}; o[d] = z; }
#pragma unroll
    for (int v = 0; v < 8; ++v) { mrun[v] = -1e30f; lrun[v] = 0.0f; }

    const int nchunks = (qblk + 64) >> 5;           // block-uniform (syncthreads-safe)
    const int qrow0   = qbase + ((lane >> 4) << 3);

    for (int kt = 0; kt < nchunks; ++kt) {
        const int kvb = kt * 32;
        v8f sc[2];
#pragma unroll
        for (int nt = 0; nt < 2; ++nt) {
            v8f s = {};
            const bf16* kp = Kk + ((size_t)(kvb + nt * 16 + (lane & 15)) * H_KV + kvh) * HD_DIM
                                + ((lane >> 4) << 3);
#pragma unroll
            for (int c = 0; c < 4; ++c) {
                v16bf kf = load_frag_b16(kp + c * 32);
                s = __builtin_amdgcn_wmma_f32_16x16x32_bf16(false, qf[c], false, kf, (short)0, s, false, false);
            }
            sc[nt] = s;
        }
        float tmax[8];
#pragma unroll
        for (int v = 0; v < 8; ++v) {
            const int q = qrow0 + v;
            float a = sc[0][v] * scale;
            float b = sc[1][v] * scale;
            if (kvb + (lane & 15) > q)      a = -1e9f;
            if (kvb + 16 + (lane & 15) > q) b = -1e9f;
            sc[0][v] = a; sc[1][v] = b;
            float m = fmaxf(a, b);
#pragma unroll
            for (int off = 8; off >= 1; off >>= 1) m = fmaxf(m, __shfl_xor(m, off, 32));
            tmax[v] = m;
        }
#pragma unroll
        for (int v = 0; v < 8; ++v) {
            const float mnew  = fmaxf(mrun[v], tmax[v]);
            const float alpha = __expf(mrun[v] - mnew);
            const float pa = __expf(sc[0][v] - mnew);
            const float pb = __expf(sc[1][v] - mnew);
            float rs = pa + pb;
#pragma unroll
            for (int off = 8; off >= 1; off >>= 1) rs += __shfl_xor(rs, off, 32);
            lrun[v] = lrun[v] * alpha + rs;
            mrun[v] = mnew;
#pragma unroll
            for (int d = 0; d < 8; ++d) o[d][v] *= alpha;
            const int prow = v + ((lane >> 4) << 3);
            P[wave][prow * 32 +      (lane & 15)] = (bf16)pa;
            P[wave][prow * 32 + 16 + (lane & 15)] = (bf16)pb;
        }
        __syncthreads();
        v16bf pf = load_frag_a(&P[wave][0], 32);
#pragma unroll
        for (int d = 0; d < 8; ++d) {
            const bf16* vp = Vt + ((size_t)(kvh * HD_DIM + d * 16 + (lane & 15))) * S_LEN
                                + kvb + ((lane >> 4) << 3);
            v16bf vf = load_frag_b16(vp);
            o[d] = __builtin_amdgcn_wmma_f32_16x16x32_bf16(false, pf, false, vf, (short)0, o[d], false, false);
        }
        __syncthreads();
    }
#pragma unroll
    for (int v = 0; v < 8; ++v) {
        const float inv = 1.0f / lrun[v];
        const int q = qrow0 + v;
        bf16* op = O + ((size_t)q * H_Q + h) * HD_DIM + (lane & 15);
#pragma unroll
        for (int d = 0; d < 8; ++d) op[d * 16] = (bf16)(o[d][v] * inv);
    }
}

// ---------------- host orchestration ----------------
static inline size_t take(size_t* off, size_t bytes) {
    size_t o = *off;
    *off = o + ((bytes + 255) & ~(size_t)255);
    return o;
}

extern "C" void kernel_launch(void* const* d_in, const int* in_sizes, int n_in,
                              void* d_out, int out_size, void* d_ws, size_t ws_size,
                              hipStream_t stream) {
    (void)in_sizes; (void)n_in; (void)out_size; (void)ws_size;
    const float* x   = (const float*)d_in[0];
    const float* wq  = (const float*)d_in[1];
    const float* wk  = (const float*)d_in[2];
    const float* wv  = (const float*)d_in[3];
    const float* wo  = (const float*)d_in[4];
    const float* qnw = (const float*)d_in[5];
    const float* knw = (const float*)d_in[6];
    const float* fc  = (const float*)d_in[9];
    const float* fs  = (const float*)d_in[10];

    char* ws = (char*)d_ws;
    size_t off = 0;
    bf16*  x_bf = (bf16*) (ws + take(&off, (size_t)S_LEN * D_DIM * 2));
    bf16*  wbig = (bf16*) (ws + take(&off, (size_t)QD * D_DIM * 2));   // wq, later wo
    bf16*  wsm  = (bf16*) (ws + take(&off, (size_t)KVD * D_DIM * 2));  // wk, later wv
    float* qf   = (float*)(ws + take(&off, (size_t)S_LEN * QD * 4));
    float* kf   = (float*)(ws + take(&off, (size_t)S_LEN * KVD * 4));
    float* vf   = (float*)(ws + take(&off, (size_t)S_LEN * KVD * 4));
    bf16*  qb   = (bf16*) (ws + take(&off, (size_t)S_LEN * QD * 2));
    bf16*  kb   = (bf16*) (ws + take(&off, (size_t)S_LEN * KVD * 2));
    bf16*  vt   = (bf16*) (ws + take(&off, (size_t)S_LEN * KVD * 2));
    bf16*  ab   = (bf16*)qf;   // alias: qf dead after norm/rope

    auto cvt = [&](const float* src, bf16* dst, size_t n) {
        size_t n4 = n / 4;
        k_cvt_bf16<<<(unsigned)((n4 + 255) / 256), 256, 0, stream>>>(src, dst, n4);
    };

    cvt(x,  x_bf, (size_t)S_LEN * D_DIM);
    cvt(wq, wbig, (size_t)QD * D_DIM);
    k_gemm_bf16<<<dim3(QD / BNg, S_LEN / BMg), 256, 0, stream>>>(x_bf, wbig, qf, S_LEN, QD, D_DIM);
    cvt(wk, wsm, (size_t)KVD * D_DIM);
    k_gemm_bf16<<<dim3(KVD / BNg, S_LEN / BMg), 256, 0, stream>>>(x_bf, wsm, kf, S_LEN, KVD, D_DIM);
    cvt(wv, wsm, (size_t)KVD * D_DIM);
    k_gemm_bf16<<<dim3(KVD / BNg, S_LEN / BMg), 256, 0, stream>>>(x_bf, wsm, vf, S_LEN, KVD, D_DIM);
    k_norm_rope<<<(S_LEN * H_Q) / 8, 256, 0, stream>>>(qf, qnw, fc, fs, qb, H_Q);
    k_norm_rope<<<(S_LEN * H_KV) / 8, 256, 0, stream>>>(kf, knw, fc, fs, kb, H_KV);
    {
        size_t n = (size_t)S_LEN * KVD;
        k_v_transpose<<<(unsigned)((n + 255) / 256), 256, 0, stream>>>(vf, vt);
    }
    k_attn<<<dim3(S_LEN / 64, H_Q), 128, 0, stream>>>(qb, kb, vt, ab);
    cvt(wo, wbig, (size_t)D_DIM * QD);
    k_gemm_bf16<<<dim3(D_DIM / BNg, S_LEN / BMg), 256, 0, stream>>>(ab, wbig, (float*)d_out,
                                                                    S_LEN, D_DIM, QD);
}
